// GAT_TimeSeries_19473381720078
// MI455X (gfx1250) — matrix operand order repaired and checked by reading, not compiled, not run on hardware
//
#include <hip/hip_runtime.h>
#include <math.h>

// ---------------------------------------------------------------------------
// Problem constants (from reference)
// ---------------------------------------------------------------------------
#define NNODES 10000
#define NEDGES 160000
#define SEQ    96
#define HC     1024     // hidden per layer (8 heads x 128)
#define OUTC   768
#define NEG_SLOPE 0.2f

// ---------------------------------------------------------------------------
// Helpers
// ---------------------------------------------------------------------------
// pack two f32 into packed bf16 pair with a single hardware instruction
__device__ __forceinline__ unsigned pkbf(float lo, float hi) {
    unsigned r;
    asm("v_cvt_pk_bf16_f32 %0, %1, %2" : "=v"(r) : "v"(lo), "v"(hi));
    return r;
}
__device__ __forceinline__ unsigned short f2bf(float f) {
    return (unsigned short)(pkbf(f, f) & 0xFFFFu);
}

// order-preserving float <-> uint key for atomicMax-based float max
__device__ __forceinline__ unsigned fkey(float f) {
    unsigned u = __float_as_uint(f);
    return (u & 0x80000000u) ? ~u : (u | 0x80000000u);
}
__device__ __forceinline__ float fdecode(unsigned k) {
    unsigned u = (k & 0x80000000u) ? (k & 0x7FFFFFFFu) : ~k;
    return __uint_as_float(u);
}

typedef __attribute__((ext_vector_type(16))) __bf16 v16bf;
typedef __attribute__((ext_vector_type(8)))  float  v8f;

union Frag16 { unsigned u[8]; v16bf v; };

// ---------------------------------------------------------------------------
// Transpose x: [SEQ, N] f32 -> xT [N, SEQ] bf16
// ---------------------------------------------------------------------------
__global__ void transpose_x(const float* __restrict__ x,
                            unsigned short* __restrict__ xT, int S, int N) {
    __shared__ float tile[32][33];
    int s0 = blockIdx.y * 32, n0 = blockIdx.x * 32;
    int s = s0 + threadIdx.y, n = n0 + threadIdx.x;
    if (s < S && n < N) tile[threadIdx.y][threadIdx.x] = x[(size_t)s * N + n];
    __syncthreads();
    int nn = n0 + threadIdx.y, ss = s0 + threadIdx.x;
    if (nn < N && ss < S) xT[(size_t)nn * S + ss] = f2bf(tile[threadIdx.x][threadIdx.y]);
}

// ---------------------------------------------------------------------------
// Pre-pack weights: W[K,N] f32 -> Wp[N][K/2] bf16-pair uints (the exact layout
// the GEMM's transposed LDS B buffer wants -> staging is a pure copy).
// ---------------------------------------------------------------------------
__global__ void pack_w(const float* __restrict__ W, unsigned* __restrict__ Wp,
                       int K, int N) {
    int idx = blockIdx.x * 256 + threadIdx.x;
    int total = N * (K >> 1);
    if (idx >= total) return;
    int n  = idx % N;          // coalesced reads across threads
    int kp = idx / N;
    float lo = W[(size_t)(2 * kp) * N + n];
    float hi = W[(size_t)(2 * kp + 1) * N + n];
    Wp[(size_t)n * (K >> 1) + kp] = pkbf(lo, hi);
}

// ---------------------------------------------------------------------------
// bf16 WMMA GEMM:  C[M,N] = A[M,K] @ B[K,N]
// A: bf16 row-major [M,K]; Bp: packed bf16-pair uints [N][K/2]; C: f32.
// Block = 256 threads (8 wave32), tile BM=128 x BN=128, BK=32.
// Double-buffered LDS, one barrier per K-chunk, zero conversion in hot loop.
// ---------------------------------------------------------------------------
#define BM 128
#define BN 128
#define BK 32

__global__ void __launch_bounds__(256)
wmma_gemm_bf16(const unsigned short* __restrict__ A, const unsigned* __restrict__ Bp,
               float* __restrict__ C, int M, int N, int K) {
    __shared__ unsigned short As[2][BM * BK];   // [row][k]
    __shared__ unsigned short Bs[2][BN * BK];   // [col][k]  (transposed)

    const int tid  = threadIdx.x;
    const int lane = tid & 31;
    const int wv   = tid >> 5;
    const int m0   = blockIdx.y * BM;
    const int n0   = blockIdx.x * BN;

    v8f acc[8];
#pragma unroll
    for (int i = 0; i < 8; ++i)
#pragma unroll
        for (int j = 0; j < 8; ++j) acc[i][j] = 0.0f;

    // A staging: thread owns 16 contiguous k of one row (2x b128)
    const int arow  = tid >> 1;            // 0..127
    const int akseg = (tid & 1) * 16;      // 0 or 16
    int grow = m0 + arow; if (grow > M - 1) grow = M - 1;   // clamp ragged M
    const unsigned short* Ag = A + (size_t)grow * K + akseg;

    // B staging: thread owns one column, 16 k values, pre-packed (2x b128)
    const int bcol = tid & 127;
    const int bks  = (tid >> 7) * 16;      // 0 or 16
    const unsigned* Bgp = Bp + (size_t)(n0 + bcol) * (K >> 1) + (bks >> 1);

    // fragment addressing (ISA 16-bit A/B layouts)
    const int mrow = (wv << 4) + (lane & 15);
    const int ak0  = (lane < 16) ? 0 : 8;
    const int ak1  = ak0 + 16;
    const int bk0  = (lane < 16) ? 0 : 8;

    const int nc = K / BK;

    // ---- prologue: fill buffer 0 ----
    {
        uint4 a0 = *(const uint4*)(Ag);
        uint4 a1 = *(const uint4*)(Ag + 8);
        uint4 b0 = *(const uint4*)(Bgp);
        uint4 b1 = *(const uint4*)(Bgp + 4);
        *(uint4*)&As[0][arow * BK + akseg]     = a0;
        *(uint4*)&As[0][arow * BK + akseg + 8] = a1;
        *(uint4*)&Bs[0][bcol * BK + bks]       = b0;
        *(uint4*)&Bs[0][bcol * BK + bks + 8]   = b1;
    }

    for (int c = 0; c < nc; ++c) {
        __syncthreads();   // LDS[c&1] ready; prior reads of LDS[(c+1)&1] done

        // issue global loads for next chunk (latency hidden by WMMAs)
        uint4 a0, a1, b0, b1;
        const bool more = (c + 1 < nc);
        if (more) {
            const int k1 = (c + 1) * BK;
            a0 = *(const uint4*)(Ag + k1);
            a1 = *(const uint4*)(Ag + k1 + 8);
            b0 = *(const uint4*)(Bgp + (k1 >> 1));
            b1 = *(const uint4*)(Bgp + (k1 >> 1) + 4);
        }

        // compute chunk c
        const int cb = c & 1;
        Frag16 af;
#pragma unroll
        for (int j = 0; j < 4; ++j) {
            af.u[j]     = *(const unsigned*)&As[cb][mrow * BK + ak0 + 2 * j];
            af.u[4 + j] = *(const unsigned*)&As[cb][mrow * BK + ak1 + 2 * j];
        }
        Frag16 bfr[8];
#pragma unroll
        for (int nt = 0; nt < 8; ++nt) {
            const int col = (nt << 4) + (lane & 15);
#pragma unroll
            for (int j = 0; j < 8; ++j)
                bfr[nt].u[j] = *(const unsigned*)&Bs[cb][col * BK + bk0 + 2 * j + ((j >> 2) << 3)];
        }
#pragma unroll
        for (int nt = 0; nt < 8; ++nt)
            acc[nt] = __builtin_amdgcn_wmma_f32_16x16x32_bf16(
                false, af.v, false, bfr[nt].v, (short)0, acc[nt], false, false);

        // store next chunk into the other buffer (pure copy)
        if (more) {
            const int nb = (c + 1) & 1;
            *(uint4*)&As[nb][arow * BK + akseg]     = a0;
            *(uint4*)&As[nb][arow * BK + akseg + 8] = a1;
            *(uint4*)&Bs[nb][bcol * BK + bks]       = b0;
            *(uint4*)&Bs[nb][bcol * BK + bks + 8]   = b1;
        }
    }

    // store: C/D layout — VGPR r: M=r (lanes 0-15) / M=8+r (lanes 16-31); N=lane%16
    const int rbase = m0 + (wv << 4) + ((lane < 16) ? 0 : 8);
    const int cbase = n0 + (lane & 15);
#pragma unroll
    for (int nt = 0; nt < 8; ++nt) {
#pragma unroll
        for (int r = 0; r < 8; ++r) {
            int row = rbase + r;
            if (row < M) C[(size_t)row * N + cbase + (nt << 4)] = acc[nt][r];
        }
    }
}

// ---------------------------------------------------------------------------
// Per-node attention coefficients: asrc[n,h] = dot(h[n,h,:], as[h,:]), same adst
// ---------------------------------------------------------------------------
template <int H, int CPT>
__global__ void __launch_bounds__(256)
attn_coeffs(const float* __restrict__ hf, const float* __restrict__ as_w,
            const float* __restrict__ ad_w, float* __restrict__ asrc,
            float* __restrict__ adst) {
    const int CT = 256 * CPT;
    const int n = blockIdx.x, t = threadIdx.x;
    __shared__ float ss[H], sd[H];
    if (t < H) { ss[t] = 0.f; sd[t] = 0.f; }
    __syncthreads();
    const int ch0 = t * CPT;
    const int hh  = ch0 / (CT / H);
    const float* row = hf + (size_t)n * CT + ch0;
    float ps = 0.f, pd = 0.f;
#pragma unroll
    for (int k = 0; k < CPT; ++k) {
        float v = row[k];
        ps += v * as_w[ch0 + k];
        pd += v * ad_w[ch0 + k];
    }
    atomicAdd(&ss[hh], ps);
    atomicAdd(&sd[hh], pd);
    __syncthreads();
    if (t < H) { asrc[(size_t)n * H + t] = ss[t]; adst[(size_t)n * H + t] = sd[t]; }
}

// ce[h] = dot(We[h*C:(h+1)*C], ae[h*C:(h+1)*C])   (rank-1 edge-feature trick)
template <int H>
__global__ void ce_kernel(const float* __restrict__ We, const float* __restrict__ ae,
                          float* __restrict__ ce, int CT) {
    __shared__ float s[H];
    int t = threadIdx.x;
    if (t < H) s[t] = 0.f;
    __syncthreads();
    const int C = CT / H;
    for (int i = t; i < CT; i += blockDim.x) atomicAdd(&s[i / C], We[i] * ae[i]);
    __syncthreads();
    if (t < H) ce[t] = s[t];
}

// ---------------------------------------------------------------------------
// CSR build by destination
// ---------------------------------------------------------------------------
__global__ void zero_ints(int* p, int n) {
    int i = blockIdx.x * 256 + threadIdx.x;
    if (i < n) p[i] = 0;
}
__global__ void copy_ints(const int* a, int* b, int n) {
    int i = blockIdx.x * 256 + threadIdx.x;
    if (i < n) b[i] = a[i];
}
__global__ void count_deg(const int* __restrict__ dst, int* cnt, int E) {
    int i = blockIdx.x * 256 + threadIdx.x;
    if (i < E) atomicAdd(&cnt[dst[i]], 1);
}
__global__ void scan_excl(const int* __restrict__ cnt, int* __restrict__ off, int n) {
    __shared__ int buf[256];
    __shared__ int carry;
    if (threadIdx.x == 0) carry = 0;
    __syncthreads();
    for (int base = 0; base < n; base += 256) {
        int i = base + threadIdx.x;
        int v = (i < n) ? cnt[i] : 0;
        buf[threadIdx.x] = v;
        __syncthreads();
        for (int s = 1; s < 256; s <<= 1) {
            int t = (threadIdx.x >= s) ? buf[threadIdx.x - s] : 0;
            __syncthreads();
            if (threadIdx.x >= s) buf[threadIdx.x] += t;
            __syncthreads();
        }
        int exc = buf[threadIdx.x] - v + carry;
        if (i < n) off[i] = exc;
        __syncthreads();
        if (threadIdx.x == 255) carry += buf[255];
        __syncthreads();
    }
    if (threadIdx.x == 0) off[n] = carry;
}
__global__ void fill_csr(const int* __restrict__ dst, int* cursor, int* eid, int E) {
    int i = blockIdx.x * 256 + threadIdx.x;
    if (i < E) {
        int pos = atomicAdd(&cursor[dst[i]], 1);
        eid[pos] = i;
    }
}

// ---------------------------------------------------------------------------
// Per-destination-node softmax attention aggregation.
// OBF=true: emit bf16 (feeds the next GEMM's A operand directly).
// ---------------------------------------------------------------------------
template <int H, int CPT, bool OBF>
__global__ void __launch_bounds__(256)
gat_aggregate(const float* __restrict__ hf, const float* __restrict__ asrc,
              const float* __restrict__ adst, const float* __restrict__ ce,
              const float* __restrict__ ew, const int* __restrict__ srcI,
              const int* __restrict__ off, const int* __restrict__ eid,
              const float* __restrict__ bias, void* __restrict__ outv, int relu) {
    const int CT = 256 * CPT;
    const int n = blockIdx.x, t = threadIdx.x;
    const int e0 = off[n], deg = off[n + 1] - e0;

    __shared__ unsigned mmax[H];
    __shared__ float ssum[H], adn[H], fm[H], invd[H];
    __shared__ float wbuf[32 * H];
    __shared__ int   sbuf[32];

    if (t < H) {
        mmax[t] = 0u;
        ssum[t] = 0.f;
        adn[t]  = adst[(size_t)n * H + t];
    }
    __syncthreads();

    // pass 1: per-head max of leaky_relu(a_src + a_dst + ew*ce)
    for (int i = t; i < deg * H; i += 256) {
        int e  = eid[e0 + i / H];
        int hh = i % H;
        float a = asrc[(size_t)srcI[e] * H + hh] + adn[hh] + ew[e] * ce[hh];
        a = (a > 0.f) ? a : NEG_SLOPE * a;
        atomicMax(&mmax[hh], fkey(a));
    }
    __syncthreads();
    if (t < H) fm[t] = (deg > 0) ? fdecode(mmax[t]) : 0.f;
    __syncthreads();

    // pass 2: per-head sum of exp
    for (int i = t; i < deg * H; i += 256) {
        int e  = eid[e0 + i / H];
        int hh = i % H;
        float a = asrc[(size_t)srcI[e] * H + hh] + adn[hh] + ew[e] * ce[hh];
        a = (a > 0.f) ? a : NEG_SLOPE * a;
        atomicAdd(&ssum[hh], __expf(a - fm[hh]));
    }
    __syncthreads();
    if (t < H) invd[t] = 1.f / (ssum[t] + 1e-16f);
    __syncthreads();

    float acc[CPT];
#pragma unroll
    for (int k = 0; k < CPT; ++k) acc[k] = 0.f;
    const int ch0 = t * CPT;
    const int myh = ch0 / (CT / H);

    for (int c0 = 0; c0 < deg; c0 += 32) {
        int cn = deg - c0; if (cn > 32) cn = 32;
        if (t < cn * H) {
            int li = t / H, hh = t % H;
            int e  = eid[e0 + c0 + li];
            int s  = srcI[e];
            float a = asrc[(size_t)s * H + hh] + adn[hh] + ew[e] * ce[hh];
            a = (a > 0.f) ? a : NEG_SLOPE * a;
            wbuf[li * H + hh] = __expf(a - fm[hh]) * invd[hh];
            if (hh == 0) sbuf[li] = s;
        }
        __syncthreads();
        for (int j = 0; j < cn; ++j) {
            float w = wbuf[j * H + myh];
            const float* row = hf + (size_t)sbuf[j] * CT + ch0;
            if constexpr (CPT == 4) {
                float4 f = *(const float4*)row;
                acc[0] += w * f.x; acc[1] += w * f.y;
                acc[2] += w * f.z; acc[3] += w * f.w;
            } else {
#pragma unroll
                for (int k = 0; k < CPT; ++k) acc[k] += w * row[k];
            }
        }
        __syncthreads();
    }

    float v[CPT];
#pragma unroll
    for (int k = 0; k < CPT; ++k) {
        v[k] = acc[k] + bias[ch0 + k];
        if (relu) v[k] = fmaxf(v[k], 0.f);
    }
    if constexpr (OBF) {
        // bf16 output (next GEMM's A operand); CPT==4 -> one 8-byte store
        unsigned short* ob = (unsigned short*)outv;
        if constexpr (CPT == 4) {
            uint2 st;
            st.x = pkbf(v[0], v[1]);
            st.y = pkbf(v[2], v[3]);
            *(uint2*)(ob + (size_t)n * CT + ch0) = st;
        } else {
#pragma unroll
            for (int k = 0; k < CPT; ++k)
                ob[(size_t)n * CT + ch0 + k] = f2bf(v[k]);
        }
    } else {
        float* of = (float*)outv;
#pragma unroll
        for (int k = 0; k < CPT; ++k) of[(size_t)n * CT + ch0 + k] = v[k];
    }
}

// ---------------------------------------------------------------------------
// Orchestration
// ---------------------------------------------------------------------------
extern "C" void kernel_launch(void* const* d_in, const int* in_sizes, int n_in,
                              void* d_out, int out_size, void* d_ws, size_t ws_size,
                              hipStream_t stream) {
    const float* x   = (const float*)d_in[0];
    const int*   ei  = (const int*)  d_in[1];
    const float* ew  = (const float*)d_in[2];
    const float* W1  = (const float*)d_in[3];
    const float* as1 = (const float*)d_in[4];
    const float* ad1 = (const float*)d_in[5];
    const float* We1 = (const float*)d_in[6];
    const float* ae1 = (const float*)d_in[7];
    const float* b1  = (const float*)d_in[8];
    const float* W2  = (const float*)d_in[9];
    const float* as2 = (const float*)d_in[10];
    const float* ad2 = (const float*)d_in[11];
    const float* We2 = (const float*)d_in[12];
    const float* ae2 = (const float*)d_in[13];
    const float* b2  = (const float*)d_in[14];
    const float* W3  = (const float*)d_in[15];
    const float* as3 = (const float*)d_in[16];
    const float* ad3 = (const float*)d_in[17];
    const float* We3 = (const float*)d_in[18];
    const float* ae3 = (const float*)d_in[19];
    const float* b3  = (const float*)d_in[20];
    float* out = (float*)d_out;

    const int Nn = NNODES, E = NEDGES;
    const int* srcI = ei;
    const int* dstI = ei + E;

    // carve workspace
    char* p = (char*)d_ws;
    auto carve = [&](size_t bytes) -> char* {
        char* r = p;
        p += (bytes + 255) & ~(size_t)255;
        return r;
    };
    unsigned short* xTb   = (unsigned short*)carve((size_t)Nn * SEQ * 2);
    float*          bufA  = (float*)carve((size_t)Nn * HC * 4);
    unsigned short* bufBb = (unsigned short*)carve((size_t)Nn * HC * 2);
    unsigned*       W1p   = (unsigned*)carve((size_t)SEQ * HC * 2);
    unsigned*       W2p   = (unsigned*)carve((size_t)HC * HC * 2);
    unsigned*       W3p   = (unsigned*)carve((size_t)HC * OUTC * 2);
    float*          asrc  = (float*)carve((size_t)Nn * 8 * 4);
    float*          adst  = (float*)carve((size_t)Nn * 8 * 4);
    float*          ce    = (float*)carve(256);
    int*            cnt   = (int*)carve((size_t)Nn * 4);
    int*            off   = (int*)carve((size_t)(Nn + 1) * 4);
    int*            cur   = (int*)carve((size_t)Nn * 4);
    int*            eid   = (int*)carve((size_t)E * 4);

    // x [1,SEQ,N] -> xT [N,SEQ] bf16
    transpose_x<<<dim3((Nn + 31) / 32, (SEQ + 31) / 32), dim3(32, 32), 0, stream>>>(
        x, xTb, SEQ, Nn);

    // pre-pack weights into bf16-pair GEMM-B layout (once per launch)
    pack_w<<<(HC * (SEQ / 2) + 255) / 256, 256, 0, stream>>>(W1, W1p, SEQ, HC);
    pack_w<<<(HC * (HC / 2) + 255) / 256, 256, 0, stream>>>(W2, W2p, HC, HC);
    pack_w<<<(OUTC * (HC / 2) + 255) / 256, 256, 0, stream>>>(W3, W3p, HC, OUTC);

    // CSR by destination (edge topology is layer-invariant)
    zero_ints<<<(Nn + 255) / 256, 256, 0, stream>>>(cnt, Nn);
    count_deg<<<(E + 255) / 256, 256, 0, stream>>>(dstI, cnt, E);
    scan_excl<<<1, 256, 0, stream>>>(cnt, off, Nn);
    copy_ints<<<(Nn + 255) / 256, 256, 0, stream>>>(off, cur, Nn);
    fill_csr<<<(E + 255) / 256, 256, 0, stream>>>(dstI, cur, eid, E);

    const dim3 gemmBlk(256);
    const int mBlocks = (Nn + BM - 1) / BM;

    // ---- Layer 1: [N,96]@[96,1024] ----
    wmma_gemm_bf16<<<dim3(HC / BN, mBlocks), gemmBlk, 0, stream>>>(xTb, W1p, bufA, Nn, HC, SEQ);
    attn_coeffs<8, 4><<<Nn, 256, 0, stream>>>(bufA, as1, ad1, asrc, adst);
    ce_kernel<8><<<1, 256, 0, stream>>>(We1, ae1, ce, HC);
    gat_aggregate<8, 4, true><<<Nn, 256, 0, stream>>>(bufA, asrc, adst, ce, ew, srcI,
                                                      off, eid, b1, bufBb, 1);

    // ---- Layer 2: [N,1024]@[1024,1024] ----
    wmma_gemm_bf16<<<dim3(HC / BN, mBlocks), gemmBlk, 0, stream>>>(bufBb, W2p, bufA, Nn, HC, HC);
    attn_coeffs<8, 4><<<Nn, 256, 0, stream>>>(bufA, as2, ad2, asrc, adst);
    ce_kernel<8><<<1, 256, 0, stream>>>(We2, ae2, ce, HC);
    gat_aggregate<8, 4, true><<<Nn, 256, 0, stream>>>(bufA, asrc, adst, ce, ew, srcI,
                                                      off, eid, b2, bufBb, 1);

    // ---- Layer 3: [N,1024]@[1024,768], 1 head, no relu, f32 out -> d_out ----
    wmma_gemm_bf16<<<dim3(OUTC / BN, mBlocks), gemmBlk, 0, stream>>>(bufBb, W3p, bufA, Nn, OUTC, HC);
    attn_coeffs<1, 3><<<Nn, 256, 0, stream>>>(bufA, as3, ad3, asrc, adst);
    ce_kernel<1><<<1, 256, 0, stream>>>(We3, ae3, ce, OUTC);
    gat_aggregate<1, 3, false><<<Nn, 256, 0, stream>>>(bufA, asrc, adst, ce, ew, srcI,
                                                       off, eid, b3, out, 0);
}
